// MoEBlock_7438883356952
// MI455X (gfx1250) — matrix operand order repaired and checked by reading, not compiled, not run on hardware
//
#include <hip/hip_runtime.h>
#include <hip/hip_bf16.h>

typedef __attribute__((ext_vector_type(16))) __bf16 v16bf;
typedef __attribute__((ext_vector_type(8)))  __bf16 v8bf;
typedef __attribute__((ext_vector_type(8)))  float  v8f;
typedef __attribute__((ext_vector_type(4)))  int    v4i;

#define HDIM 1024
#define IDIM 1024
#define NEXP 8
#define NTOK 8192
#define BM 128
#define BN 128
#define BK 32
#define NT (HDIM / BK)
#define LDP 40   // padded LDS row stride (bf16): 80B rows -> aligned b128, 20-bank stride

#define GLOBAL_AS __attribute__((address_space(1)))
#define LDS_AS    __attribute__((address_space(3)))

#if __has_builtin(__builtin_amdgcn_global_load_async_to_lds_b128)
#define ASYNC_OK 1
#else
#define ASYNC_OK 0
#endif

#if ASYNC_OK
#if __has_builtin(__builtin_amdgcn_s_wait_asynccnt)
#define WAIT_ASYNC(n) __builtin_amdgcn_s_wait_asynccnt(n)
#else
#define WAIT_ASYNC(n) asm volatile("s_wait_asynccnt %0" ::"i"(n))
#endif
#else
#define WAIT_ASYNC(n) do { } while (0)
#endif

// 16-byte global -> LDS copy: async on gfx1250 if available, sync fallback otherwise.
__device__ __forceinline__ void copy16(const __bf16* g, __bf16* l) {
#if ASYNC_OK
  __builtin_amdgcn_global_load_async_to_lds_b128((GLOBAL_AS v4i*)g, (LDS_AS v4i*)l, 0, 0);
#else
  *(v8bf*)l = *(const v8bf*)g;
#endif
}

// ---------------- one-time converts: x -> bf16; W -> bf16 transposed [E][N][K] ----------------
__global__ __launch_bounds__(256) void convert_x_kernel(const float* __restrict__ x,
                                                        __bf16* __restrict__ xbf) {
  const size_t i = ((size_t)blockIdx.x * 256 + threadIdx.x) * 8;
  const float4 a = *(const float4*)(x + i);
  const float4 b = *(const float4*)(x + i + 4);
  v8bf t;
  t[0] = (__bf16)a.x; t[1] = (__bf16)a.y; t[2] = (__bf16)a.z; t[3] = (__bf16)a.w;
  t[4] = (__bf16)b.x; t[5] = (__bf16)b.y; t[6] = (__bf16)b.z; t[7] = (__bf16)b.w;
  *(v8bf*)(xbf + i) = t;
}

__global__ __launch_bounds__(256) void convert_wT_kernel(const float* __restrict__ W,
                                                         __bf16* __restrict__ WT) {
  // W: [E][K][N] f32 row-major -> WT: [E][N][K] bf16
  const int e  = blockIdx.z;
  const int k0 = blockIdx.x * 64;
  const int n0 = blockIdx.y * 64;
  __shared__ __bf16 tile[64 * 72];  // [n][k], 144B row stride (16B aligned)
  const int tid = threadIdx.x;
  {
    const int n4 = (tid & 15) * 4;
    int k = tid >> 4;  // 0..15
#pragma unroll
    for (int it = 0; it < 4; ++it, k += 16) {
      const float4 v = *(const float4*)(W + ((size_t)e * 1024 + k0 + k) * 1024 + n0 + n4);
      tile[(n4 + 0) * 72 + k] = (__bf16)v.x;
      tile[(n4 + 1) * 72 + k] = (__bf16)v.y;
      tile[(n4 + 2) * 72 + k] = (__bf16)v.z;
      tile[(n4 + 3) * 72 + k] = (__bf16)v.w;
    }
  }
  __syncthreads();
#pragma unroll
  for (int it = 0; it < 2; ++it) {
    const int ch = tid + it * 256;  // 512 chunks of 16B
    const int n = ch >> 3;
    const int c = ch & 7;
    const v8bf val = *(const v8bf*)&tile[n * 72 + c * 8];
    *(v8bf*)(WT + ((size_t)e * 1024 + n0 + n) * 1024 + k0 + c * 8) = val;
  }
}

// ---------------- gating: logits + argmax (softmax is monotone) ----------------
__global__ __launch_bounds__(256) void gate_kernel(const float* __restrict__ x,
                                                   const float* __restrict__ Wg,
                                                   const float* __restrict__ bg,
                                                   int* __restrict__ idx) {
  const int wid  = threadIdx.x >> 5;
  const int lane = threadIdx.x & 31;
  const int t    = blockIdx.x * 8 + wid;
  float acc[NEXP];
#pragma unroll
  for (int e = 0; e < NEXP; ++e) acc[e] = 0.f;
  const float* xrow = x + (size_t)t * HDIM;
#pragma unroll 4
  for (int j = 0; j < HDIM / 32; ++j) {
    const int h = lane + j * 32;
    const float xv = xrow[h];
    const float4 w0 = *(const float4*)(Wg + h * NEXP);
    const float4 w1 = *(const float4*)(Wg + h * NEXP + 4);
    acc[0] += xv * w0.x; acc[1] += xv * w0.y; acc[2] += xv * w0.z; acc[3] += xv * w0.w;
    acc[4] += xv * w1.x; acc[5] += xv * w1.y; acc[6] += xv * w1.z; acc[7] += xv * w1.w;
  }
#pragma unroll
  for (int e = 0; e < NEXP; ++e) {
#pragma unroll
    for (int off = 16; off > 0; off >>= 1)
      acc[e] += __shfl_xor(acc[e], off, 32);
  }
  if (lane == 0) {
    int best = 0;
    float bv = acc[0] + bg[0];
#pragma unroll
    for (int e = 1; e < NEXP; ++e) {
      const float v = acc[e] + bg[e];
      if (v > bv) { bv = v; best = e; }
    }
    idx[t] = best;
  }
}

// ---------------- routing ----------------
__global__ void zero_counts_kernel(int* __restrict__ counts) {
  if (threadIdx.x < NEXP) counts[threadIdx.x] = 0;
}
__global__ void count_kernel(const int* __restrict__ idx, int* __restrict__ counts,
                             int* __restrict__ pos) {
  const int t = blockIdx.x * 256 + threadIdx.x;
  if (t < NTOK) pos[t] = atomicAdd(&counts[idx[t]], 1);
}
__global__ void offsets_kernel(const int* __restrict__ counts, int* __restrict__ offsets) {
  if (threadIdx.x == 0) {
    int s = 0;
    for (int e = 0; e < NEXP; ++e) { offsets[e] = s; s += counts[e]; }
  }
}
__global__ void scatter_kernel(const int* __restrict__ idx, const int* __restrict__ pos,
                               const int* __restrict__ offsets, int* __restrict__ sorted) {
  const int t = blockIdx.x * 256 + threadIdx.x;
  if (t < NTOK) sorted[offsets[idx[t]] + pos[t]] = t;
}

// ---------------- FFN pass 1: h = gelu(xbf_gathered @ W1T^T + b1) -> bf16 ----------------
__global__ __launch_bounds__(256) void ffn1_kernel(const __bf16* __restrict__ xbf,
                                                   const __bf16* __restrict__ w1t,
                                                   const float* __restrict__ b1,
                                                   const int* __restrict__ sorted,
                                                   const int* __restrict__ counts,
                                                   const int* __restrict__ offsets,
                                                   __bf16* __restrict__ hbuf) {
  const int e   = blockIdx.z;
  const int cnt = counts[e];
  const int m0  = blockIdx.x * BM;
  if (m0 >= cnt) return;
  const int n0   = blockIdx.y * BN;
  const int offs = offsets[e];

  __shared__ __bf16 As0[BM * LDP], As1[BM * LDP];
  __shared__ __bf16 Bs0[BN * LDP], Bs1[BN * LDP];
  __shared__ int    toks[BM];

  const int tid = threadIdx.x;
  if (tid < BM) toks[tid] = sorted[offs + min(m0 + tid, cnt - 1)];
  __syncthreads();

  // per-thread copy chunks: rows rA and rA+64, 16B sub-chunk cc
  const int rA = tid >> 2;
  const int cc = (tid & 3) * 8;
  const int d0 = rA * LDP + cc;
  const int d1 = (rA + 64) * LDP + cc;
  const __bf16* srcA0 = xbf + (size_t)toks[rA] * HDIM + cc;
  const __bf16* srcA1 = xbf + (size_t)toks[rA + 64] * HDIM + cc;
  const __bf16* srcB0 = w1t + ((size_t)e * IDIM + n0 + rA) * HDIM + cc;
  const __bf16* srcB1 = w1t + ((size_t)e * IDIM + n0 + rA + 64) * HDIM + cc;

  const int wid = tid >> 5, lane = tid & 31, lrow = lane & 15, lhi = lane >> 4;
  const int waveM = wid & 3, waveN = wid >> 2;

  v8f acc[2][4];
#pragma unroll
  for (int mi = 0; mi < 2; ++mi)
#pragma unroll
    for (int ni = 0; ni < 4; ++ni)
#pragma unroll
      for (int r = 0; r < 8; ++r) acc[mi][ni][r] = 0.f;

  // prologue: tile 0 into buffer 0
  copy16(srcA0, As0 + d0); copy16(srcA1, As0 + d1);
  copy16(srcB0, Bs0 + d0); copy16(srcB1, Bs0 + d1);

  for (int it = 0; it < NT; ++it) {
    if (it + 1 < NT) {
      const int kn = (it + 1) * BK;
      __bf16* Ad = (it & 1) ? As0 : As1;
      __bf16* Bd = (it & 1) ? Bs0 : Bs1;
      copy16(srcA0 + kn, Ad + d0); copy16(srcA1 + kn, Ad + d1);
      copy16(srcB0 + kn, Bd + d0); copy16(srcB1 + kn, Bd + d1);
      WAIT_ASYNC(4);
    } else {
      WAIT_ASYNC(0);
    }
    __syncthreads();
    const __bf16* Ab = (it & 1) ? As1 : As0;
    const __bf16* Bb = (it & 1) ? Bs1 : Bs0;

    union { v16bf v; v8bf h[2]; } a[2], b[4];
#pragma unroll
    for (int mi = 0; mi < 2; ++mi) {
      const __bf16* p = &Ab[(waveM * 32 + mi * 16 + lrow) * LDP + lhi * 8];
      a[mi].h[0] = *(const v8bf*)p;         // K = kb .. kb+7
      a[mi].h[1] = *(const v8bf*)(p + 16);  // K = kb+16 .. kb+23
    }
#pragma unroll
    for (int ni = 0; ni < 4; ++ni) {
      const __bf16* p = &Bb[(waveN * 64 + ni * 16 + lrow) * LDP + lhi * 8];
      b[ni].h[0] = *(const v8bf*)p;         // K = kb .. kb+7
      b[ni].h[1] = *(const v8bf*)(p + 16);  // K = kb+16 .. kb+23
    }
#pragma unroll
    for (int mi = 0; mi < 2; ++mi)
#pragma unroll
      for (int ni = 0; ni < 4; ++ni)
        acc[mi][ni] = __builtin_amdgcn_wmma_f32_16x16x32_bf16(
            false, a[mi].v, false, b[ni].v, (short)0, acc[mi][ni], false, false);
    __syncthreads();
  }

#pragma unroll
  for (int mi = 0; mi < 2; ++mi) {
#pragma unroll
    for (int ni = 0; ni < 4; ++ni) {
      const int col = n0 + waveN * 64 + ni * 16 + lrow;
      const float bv = b1[e * IDIM + col];
#pragma unroll
      for (int r = 0; r < 8; ++r) {
        const int row = waveM * 32 + mi * 16 + lhi * 8 + r;
        const int gm = m0 + row;
        if (gm < cnt) {
          const float v = acc[mi][ni][r] + bv;
          const float g = 0.5f * v * (1.0f + erff(v * 0.70710678118654752f));
          hbuf[(size_t)(offs + gm) * IDIM + col] = (__bf16)g;
        }
      }
    }
  }
}

// ---------------- FFN pass 2: out = h @ W2T^T + b2, scatter to token rows ----------------
__global__ __launch_bounds__(256) void ffn2_kernel(const __bf16* __restrict__ hbuf,
                                                   const __bf16* __restrict__ w2t,
                                                   const float* __restrict__ b2,
                                                   const int* __restrict__ sorted,
                                                   const int* __restrict__ counts,
                                                   const int* __restrict__ offsets,
                                                   float* __restrict__ out) {
  const int e   = blockIdx.z;
  const int cnt = counts[e];
  const int m0  = blockIdx.x * BM;
  if (m0 >= cnt) return;
  const int n0   = blockIdx.y * BN;
  const int offs = offsets[e];

  __shared__ __bf16 As0[BM * LDP], As1[BM * LDP];
  __shared__ __bf16 Bs0[BN * LDP], Bs1[BN * LDP];
  __shared__ int    toks[BM];

  const int tid = threadIdx.x;
  if (tid < BM) toks[tid] = sorted[offs + min(m0 + tid, cnt - 1)];
  __syncthreads();

  const int rA = tid >> 2;
  const int cc = (tid & 3) * 8;
  const int d0 = rA * LDP + cc;
  const int d1 = (rA + 64) * LDP + cc;
  const __bf16* srcA0 = hbuf + (size_t)(offs + min(m0 + rA, cnt - 1)) * IDIM + cc;
  const __bf16* srcA1 = hbuf + (size_t)(offs + min(m0 + rA + 64, cnt - 1)) * IDIM + cc;
  const __bf16* srcB0 = w2t + ((size_t)e * HDIM + n0 + rA) * IDIM + cc;
  const __bf16* srcB1 = w2t + ((size_t)e * HDIM + n0 + rA + 64) * IDIM + cc;

  const int wid = tid >> 5, lane = tid & 31, lrow = lane & 15, lhi = lane >> 4;
  const int waveM = wid & 3, waveN = wid >> 2;

  v8f acc[2][4];
#pragma unroll
  for (int mi = 0; mi < 2; ++mi)
#pragma unroll
    for (int ni = 0; ni < 4; ++ni)
#pragma unroll
      for (int r = 0; r < 8; ++r) acc[mi][ni][r] = 0.f;

  copy16(srcA0, As0 + d0); copy16(srcA1, As0 + d1);
  copy16(srcB0, Bs0 + d0); copy16(srcB1, Bs0 + d1);

  for (int it = 0; it < NT; ++it) {
    if (it + 1 < NT) {
      const int kn = (it + 1) * BK;
      __bf16* Ad = (it & 1) ? As0 : As1;
      __bf16* Bd = (it & 1) ? Bs0 : Bs1;
      copy16(srcA0 + kn, Ad + d0); copy16(srcA1 + kn, Ad + d1);
      copy16(srcB0 + kn, Bd + d0); copy16(srcB1 + kn, Bd + d1);
      WAIT_ASYNC(4);
    } else {
      WAIT_ASYNC(0);
    }
    __syncthreads();
    const __bf16* Ab = (it & 1) ? As1 : As0;
    const __bf16* Bb = (it & 1) ? Bs1 : Bs0;

    union { v16bf v; v8bf h[2]; } a[2], b[4];
#pragma unroll
    for (int mi = 0; mi < 2; ++mi) {
      const __bf16* p = &Ab[(waveM * 32 + mi * 16 + lrow) * LDP + lhi * 8];
      a[mi].h[0] = *(const v8bf*)p;
      a[mi].h[1] = *(const v8bf*)(p + 16);
    }
#pragma unroll
    for (int ni = 0; ni < 4; ++ni) {
      const __bf16* p = &Bb[(waveN * 64 + ni * 16 + lrow) * LDP + lhi * 8];
      b[ni].h[0] = *(const v8bf*)p;
      b[ni].h[1] = *(const v8bf*)(p + 16);
    }
#pragma unroll
    for (int mi = 0; mi < 2; ++mi)
#pragma unroll
      for (int ni = 0; ni < 4; ++ni)
        acc[mi][ni] = __builtin_amdgcn_wmma_f32_16x16x32_bf16(
            false, a[mi].v, false, b[ni].v, (short)0, acc[mi][ni], false, false);
    __syncthreads();
  }

#pragma unroll
  for (int mi = 0; mi < 2; ++mi) {
#pragma unroll
    for (int ni = 0; ni < 4; ++ni) {
      const int col = n0 + waveN * 64 + ni * 16 + lrow;
      const float bv = b2[e * HDIM + col];
#pragma unroll
      for (int r = 0; r < 8; ++r) {
        const int row = waveM * 32 + mi * 16 + lhi * 8 + r;
        const int gm = m0 + row;
        if (gm < cnt) out[(size_t)toks[row] * HDIM + col] = acc[mi][ni][r] + bv;
      }
    }
  }
}

extern "C" void kernel_launch(void* const* d_in, const int* in_sizes, int n_in,
                              void* d_out, int out_size, void* d_ws, size_t ws_size,
                              hipStream_t stream) {
  const float* x  = (const float*)d_in[0];
  const float* Wg = (const float*)d_in[1];
  const float* bg = (const float*)d_in[2];
  const float* W1 = (const float*)d_in[3];
  const float* b1 = (const float*)d_in[4];
  const float* W2 = (const float*)d_in[5];
  const float* b2 = (const float*)d_in[6];
  float* out = (float*)d_out;

  char* ws = (char*)d_ws;
  int* idx     = (int*)(ws);
  int* pos     = (int*)(ws + 4 * NTOK);
  int* counts  = (int*)(ws + 8 * NTOK);
  int* offsets = (int*)(ws + 8 * NTOK + 64);
  int* sorted  = (int*)(ws + 8 * NTOK + 128);
  size_t o = 131072;
  __bf16* xbf  = (__bf16*)(ws + o); o += (size_t)NTOK * HDIM * 2;
  __bf16* hbuf = (__bf16*)(ws + o); o += (size_t)NTOK * IDIM * 2;
  __bf16* w1t  = (__bf16*)(ws + o); o += (size_t)NEXP * HDIM * IDIM * 2;
  __bf16* w2t  = (__bf16*)(ws + o);

  convert_x_kernel<<<(NTOK * HDIM) / 2048, 256, 0, stream>>>(x, xbf);
  convert_wT_kernel<<<dim3(16, 16, NEXP), 256, 0, stream>>>(W1, w1t);
  convert_wT_kernel<<<dim3(16, 16, NEXP), 256, 0, stream>>>(W2, w2t);

  gate_kernel<<<NTOK / 8, 256, 0, stream>>>(x, Wg, bg, idx);
  zero_counts_kernel<<<1, 32, 0, stream>>>(counts);
  count_kernel<<<NTOK / 256, 256, 0, stream>>>(idx, counts, pos);
  offsets_kernel<<<1, 32, 0, stream>>>(counts, offsets);
  scatter_kernel<<<NTOK / 256, 256, 0, stream>>>(idx, pos, offsets, sorted);

  dim3 g1(NTOK / BM, IDIM / BN, NEXP);
  ffn1_kernel<<<g1, 256, 0, stream>>>(xbf, w1t, b1, sorted, counts, offsets, hbuf);
  dim3 g2(NTOK / BM, HDIM / BN, NEXP);
  ffn2_kernel<<<g2, 256, 0, stream>>>(hbuf, w2t, b2, sorted, counts, offsets, out);
}
